// NOVA_GNN_41059887350180
// MI455X (gfx1250) — matrix operand order repaired and checked by reading, not compiled, not run on hardware
//
#include <hip/hip_runtime.h>
#include <hip/hip_bf16.h>

// ---------------------------------------------------------------------------
// Hetero-SAGE GNN for MI455X (gfx1250, wave32).
// Dense blocks use V_WMMA_F32_16X16X32_BF16 (bf16 A/B, f32 accum).
// Weights are staged into LDS pre-swizzled into WMMA B-fragment order so each
// lane's fragment is a contiguous 32B v16bf -> ds_load_b128 pairs, not 16x
// ds_load_u16. A fragments load as 4x float4 then convert (loads batched).
// Edge aggregation: per-(edge,feature) global f32 atomics (L2-resident).
// ---------------------------------------------------------------------------

typedef __attribute__((ext_vector_type(16))) __bf16 v16bf;
typedef __attribute__((ext_vector_type(8)))  float  v8f;

#define GNN_H      64
#define GNN_USERS  100000
#define GNN_MOVIES 20000
#define GNN_FEAT   256
#define GNN_E      2000000
#define GNN_EL     500000

// ---- A fragment: 16x32 bf16 tile row, scaled, from f32 ---------------------
// Lane (lo,hi) needs k = {hi*8 .. hi*8+7} and {16+hi*8 .. 16+hi*8+7} relative
// to kbase: two contiguous 8-float runs -> 4x float4 loads, then convert.
__device__ __forceinline__ v16bf gnn_load_a_f32(const float* __restrict__ X,
                                                int ld, int row, int kbase,
                                                int hi, float scale) {
  const float* xr = X + (size_t)row * ld + kbase + hi * 8;
  float4 p0 = *(const float4*)(xr);
  float4 p1 = *(const float4*)(xr + 4);
  float4 p2 = *(const float4*)(xr + 16);
  float4 p3 = *(const float4*)(xr + 20);
  v16bf a;
  a[0]  = (__bf16)(p0.x * scale); a[1]  = (__bf16)(p0.y * scale);
  a[2]  = (__bf16)(p0.z * scale); a[3]  = (__bf16)(p0.w * scale);
  a[4]  = (__bf16)(p1.x * scale); a[5]  = (__bf16)(p1.y * scale);
  a[6]  = (__bf16)(p1.z * scale); a[7]  = (__bf16)(p1.w * scale);
  a[8]  = (__bf16)(p2.x * scale); a[9]  = (__bf16)(p2.y * scale);
  a[10] = (__bf16)(p2.z * scale); a[11] = (__bf16)(p2.w * scale);
  a[12] = (__bf16)(p3.x * scale); a[13] = (__bf16)(p3.y * scale);
  a[14] = (__bf16)(p3.z * scale); a[15] = (__bf16)(p3.w * scale);
  return a;
}

// ---- stage W (K x 64 f32, row-major) into LDS in B-fragment order ----------
// Fragment (kk,t) lane L element e2 corresponds to W[kk*32 + (e2>>3)*16 +
// (L>>4)*8 + (e2&7)][t*16 + (L&15)]. LDS writes are contiguous per thread; global reads hit L2.
__device__ __forceinline__ void gnn_stage_w_frag(const float* __restrict__ W,
                                                 v16bf* __restrict__ sFrag,
                                                 int K, int tid, int nthreads) {
  __bf16* flatp = (__bf16*)sFrag;
  int total = K * GNN_H;  // == number of bf16 elements
  for (int flat = tid; flat < total; flat += nthreads) {
    int e2   = flat & 15;
    int frag = flat >> 4;
    int lane = frag & 31;
    int ft   = frag >> 5;
    int t    = ft & 3;
    int kk   = ft >> 2;
    int hi   = lane >> 4, lo = lane & 15;
    int k = kk * 32 + (e2 >> 3) * 16 + hi * 8 + (e2 & 7);
    int n = t * 16 + lo;
    flatp[flat] = (__bf16)W[k * GNN_H + n];
  }
}

// ---- utility kernels -------------------------------------------------------
__global__ void gnn_zero_kernel(float* __restrict__ p, int n) {
  int i = blockIdx.x * blockDim.x + threadIdx.x;
  if (i < n) p[i] = 0.0f;
}

__global__ void gnn_count_kernel(const int* __restrict__ idx,
                                 float* __restrict__ cnt, int n) {
  int e = blockIdx.x * blockDim.x + threadIdx.x;
  if (e < n) atomicAdd(&cnt[idx[e]], 1.0f);
}

// one thread per (edge, feature): coalesced gather + parallel atomic scatter
__global__ void gnn_scatter_add_kernel(const float* __restrict__ x,
                                       const int* __restrict__ srcI,
                                       const int* __restrict__ dstI,
                                       float* __restrict__ agg, int nedge) {
  long long t = (long long)blockIdx.x * blockDim.x + threadIdx.x;
  int e = (int)(t >> 6);
  int h = (int)(t & 63);
  if (e < nedge) {
    float v = x[(size_t)srcI[e] * GNN_H + h];
    atomicAdd(&agg[(size_t)dstI[e] * GNN_H + h], v);
  }
}

// one wave per row; H=64 -> 2 values per lane, shfl_xor reduction
__global__ void gnn_l2norm_kernel(float* __restrict__ x, int nrows) {
  int gt = blockIdx.x * blockDim.x + threadIdx.x;
  int wave = gt >> 5;
  int lane = gt & 31;
  if (wave >= nrows) return;
  float a = x[wave * GNN_H + lane];
  float b = x[wave * GNN_H + 32 + lane];
  float s = a * a + b * b;
#pragma unroll
  for (int off = 16; off >= 1; off >>= 1) s += __shfl_xor(s, off, 32);
  float inv = 1.0f / fmaxf(sqrtf(s), 1e-12f);
  x[wave * GNN_H + lane]      = a * inv;
  x[wave * GNN_H + 32 + lane] = b * inv;
}

// ---- projection: movie0 = relu(movie_x @ proj_w + proj_b) ------------------
__global__ void gnn_proj_relu_kernel(const float* __restrict__ movie_x,
                                     const float* __restrict__ proj_w,
                                     const float* __restrict__ proj_b,
                                     float* __restrict__ out, int ntiles) {
  __shared__ v16bf sW[(GNN_FEAT / 32) * 4 * 32];  // 1024 frags = 32 KB
  __shared__ float sB[GNN_H];
  int tid = threadIdx.x;
  gnn_stage_w_frag(proj_w, sW, GNN_FEAT, tid, blockDim.x);
  if (tid < GNN_H) sB[tid] = proj_b[tid];
  __syncthreads();

  int tile = blockIdx.x * (blockDim.x >> 5) + (tid >> 5);
  if (tile >= ntiles) return;
  int lane = tid & 31, lo = lane & 15, hi = lane >> 4;
  int row0 = tile * 16;

  v8f acc[4];
#pragma unroll
  for (int t = 0; t < 4; ++t) {
    float bv = sB[t * 16 + lo];
#pragma unroll
    for (int r = 0; r < 8; ++r) acc[t][r] = bv;
  }
#pragma unroll
  for (int kk = 0; kk < GNN_FEAT / 32; ++kk) {
    v16bf a = gnn_load_a_f32(movie_x, GNN_FEAT, row0 + lo, kk * 32, hi, 1.0f);
#pragma unroll
    for (int t = 0; t < 4; ++t) {
      v16bf b = sW[((kk * 4 + t) << 5) + lane];
      acc[t] = __builtin_amdgcn_wmma_f32_16x16x32_bf16(
          false, a, false, b, (short)0, acc[t], false, false);
    }
  }
#pragma unroll
  for (int t = 0; t < 4; ++t)
#pragma unroll
    for (int r = 0; r < 8; ++r) {
      int row = row0 + hi * 8 + r;
      out[row * GNN_H + t * 16 + lo] = fmaxf(acc[t][r], 0.0f);
    }
}

// ---- SAGE transform (mean division folded into A load as row scale):
// out = [relu_res ? xdst + relu : id]( (agg/cnt) @ wl + bl + xdst @ wr )
__global__ void gnn_sage_linear_kernel(const float* __restrict__ agg,
                                       const float* __restrict__ cnt,
                                       const float* __restrict__ xdst,
                                       const float* __restrict__ wl,
                                       const float* __restrict__ bl,
                                       const float* __restrict__ wr,
                                       float* __restrict__ out,
                                       int ntiles, int relu_residual) {
  __shared__ v16bf sWl[2 * 4 * 32];  // 256 frags = 8 KB
  __shared__ v16bf sWr[2 * 4 * 32];  // 256 frags = 8 KB
  __shared__ float sB[GNN_H];
  int tid = threadIdx.x;
  gnn_stage_w_frag(wl, sWl, GNN_H, tid, blockDim.x);
  gnn_stage_w_frag(wr, sWr, GNN_H, tid, blockDim.x);
  if (tid < GNN_H) sB[tid] = bl[tid];
  __syncthreads();

  int tile = blockIdx.x * (blockDim.x >> 5) + (tid >> 5);
  if (tile >= ntiles) return;
  int lane = tid & 31, lo = lane & 15, hi = lane >> 4;
  int row0 = tile * 16;
  float inv = 1.0f / fmaxf(cnt[row0 + lo], 1.0f);  // per-lane row scale

  v8f acc[4];
#pragma unroll
  for (int t = 0; t < 4; ++t) {
    float bv = sB[t * 16 + lo];
#pragma unroll
    for (int r = 0; r < 8; ++r) acc[t][r] = bv;
  }
#pragma unroll
  for (int kk = 0; kk < 2; ++kk) {
    v16bf a = gnn_load_a_f32(agg, GNN_H, row0 + lo, kk * 32, hi, inv);
#pragma unroll
    for (int t = 0; t < 4; ++t) {
      v16bf b = sWl[((kk * 4 + t) << 5) + lane];
      acc[t] = __builtin_amdgcn_wmma_f32_16x16x32_bf16(
          false, a, false, b, (short)0, acc[t], false, false);
    }
  }
#pragma unroll
  for (int kk = 0; kk < 2; ++kk) {
    v16bf a = gnn_load_a_f32(xdst, GNN_H, row0 + lo, kk * 32, hi, 1.0f);
#pragma unroll
    for (int t = 0; t < 4; ++t) {
      v16bf b = sWr[((kk * 4 + t) << 5) + lane];
      acc[t] = __builtin_amdgcn_wmma_f32_16x16x32_bf16(
          false, a, false, b, (short)0, acc[t], false, false);
    }
  }
#pragma unroll
  for (int t = 0; t < 4; ++t)
#pragma unroll
    for (int r = 0; r < 8; ++r) {
      int row = row0 + hi * 8 + r;
      int col = t * 16 + lo;
      float v = acc[t][r];
      if (relu_residual) v = xdst[row * GNN_H + col] + fmaxf(v, 0.0f);
      out[row * GNN_H + col] = v;
    }
}

// ---- decoder: relu(concat(u[ls],m[ld]) @ w1 + b1) @ w2 + b2 ----------------
__global__ void gnn_decoder_kernel(const float* __restrict__ u,
                                   const float* __restrict__ m,
                                   const int* __restrict__ ls,
                                   const int* __restrict__ ld,
                                   const float* __restrict__ w1,
                                   const float* __restrict__ b1,
                                   const float* __restrict__ w2,
                                   const float* __restrict__ b2,
                                   float* __restrict__ out, int ntiles) {
  __shared__ v16bf sW1[4 * 4 * 32];  // 512 frags = 16 KB (K = 128)
  __shared__ float sB1[GNN_H];
  __shared__ float sW2[GNN_H];
  int tid = threadIdx.x;
  gnn_stage_w_frag(w1, sW1, 2 * GNN_H, tid, blockDim.x);
  if (tid < GNN_H) { sB1[tid] = b1[tid]; sW2[tid] = w2[tid]; }
  __syncthreads();

  int tile = blockIdx.x * (blockDim.x >> 5) + (tid >> 5);
  if (tile >= ntiles) return;
  int lane = tid & 31, lo = lane & 15, hi = lane >> 4;
  int i0 = tile * 16;
  int ur = ls[i0 + lo];
  int mr = ld[i0 + lo];

  v8f acc[4];
#pragma unroll
  for (int t = 0; t < 4; ++t) {
    float bv = sB1[t * 16 + lo];
#pragma unroll
    for (int r = 0; r < 8; ++r) acc[t][r] = bv;
  }
#pragma unroll
  for (int kk = 0; kk < 4; ++kk) {
    v16bf a = (kk < 2) ? gnn_load_a_f32(u, GNN_H, ur, kk * 32, hi, 1.0f)
                       : gnn_load_a_f32(m, GNN_H, mr, (kk - 2) * 32, hi, 1.0f);
#pragma unroll
    for (int t = 0; t < 4; ++t) {
      v16bf b = sW1[((kk * 4 + t) << 5) + lane];
      acc[t] = __builtin_amdgcn_wmma_f32_16x16x32_bf16(
          false, a, false, b, (short)0, acc[t], false, false);
    }
  }
  float b2v = b2[0];
#pragma unroll
  for (int r = 0; r < 8; ++r) {
    float s = 0.0f;
#pragma unroll
    for (int t = 0; t < 4; ++t)
      s += fmaxf(acc[t][r], 0.0f) * sW2[t * 16 + lo];
    // reduce over the 16 lanes of this half-wave (N dimension)
#pragma unroll
    for (int off = 8; off >= 1; off >>= 1) s += __shfl_xor(s, off, 32);
    if (lo == 0) out[i0 + hi * 8 + r] = s + b2v;
  }
}

// ---------------------------------------------------------------------------
extern "C" void kernel_launch(void* const* d_in, const int* in_sizes, int n_in,
                              void* d_out, int out_size, void* d_ws, size_t ws_size,
                              hipStream_t stream) {
  (void)in_sizes; (void)n_in; (void)out_size; (void)ws_size;

  const float* movie_x  = (const float*)d_in[0];
  const float* user_emb = (const float*)d_in[1];
  const float* proj_w   = (const float*)d_in[2];
  const float* proj_b   = (const float*)d_in[3];
  const float* c1_um_wl = (const float*)d_in[4];
  const float* c1_um_bl = (const float*)d_in[5];
  const float* c1_um_wr = (const float*)d_in[6];
  const float* c1_mu_wl = (const float*)d_in[7];
  const float* c1_mu_bl = (const float*)d_in[8];
  const float* c1_mu_wr = (const float*)d_in[9];
  const float* c2_um_wl = (const float*)d_in[10];
  const float* c2_um_bl = (const float*)d_in[11];
  const float* c2_um_wr = (const float*)d_in[12];
  const float* c2_mu_wl = (const float*)d_in[13];
  const float* c2_mu_bl = (const float*)d_in[14];
  const float* c2_mu_wr = (const float*)d_in[15];
  const float* dec_w1   = (const float*)d_in[16];
  const float* dec_b1   = (const float*)d_in[17];
  const float* dec_w2   = (const float*)d_in[18];
  const float* dec_b2   = (const float*)d_in[19];
  const int*   edge_src = (const int*)d_in[20];
  const int*   edge_dst = (const int*)d_in[21];
  const int*   lbl_src  = (const int*)d_in[22];
  const int*   lbl_dst  = (const int*)d_in[23];
  float* out = (float*)d_out;

  // workspace carve-up (~98 MB of f32)
  float* base = (float*)d_ws;
  size_t off = 0;
  float* movie0 = base + off; off += (size_t)GNN_MOVIES * GNN_H;
  float* agg_m  = base + off; off += (size_t)GNN_MOVIES * GNN_H;
  float* agg_u  = base + off; off += (size_t)GNN_USERS  * GNN_H;
  float* cnt_m  = base + off; off += GNN_MOVIES;
  float* cnt_u  = base + off; off += GNN_USERS;
  float* m_res  = base + off; off += (size_t)GNN_MOVIES * GNN_H;
  float* u_res  = base + off; off += (size_t)GNN_USERS  * GNN_H;
  float* m2     = base + off; off += (size_t)GNN_MOVIES * GNN_H;
  float* u2     = base + off; off += (size_t)GNN_USERS  * GNN_H;

  const int B = 256;
  int mt = GNN_MOVIES / 16;    // 1250 row tiles
  int ut = GNN_USERS / 16;     // 6250 row tiles
  int lt = GNN_EL / 16;        // 31250 row tiles
  int cgrid = (GNN_E + B - 1) / B;
  long long sthreads = (long long)GNN_E * 64;
  int sgrid = (int)((sthreads + B - 1) / B);
  int mN = GNN_MOVIES * GNN_H, uN = GNN_USERS * GNN_H;
  int dmg = (mN + B - 1) / B, dug = (uN + B - 1) / B;

  // neighbor counts (shared by both layers)
  gnn_zero_kernel<<<(GNN_MOVIES + B - 1) / B, B, 0, stream>>>(cnt_m, GNN_MOVIES);
  gnn_zero_kernel<<<(GNN_USERS + B - 1) / B, B, 0, stream>>>(cnt_u, GNN_USERS);
  gnn_count_kernel<<<cgrid, B, 0, stream>>>(edge_dst, cnt_m, GNN_E);
  gnn_count_kernel<<<cgrid, B, 0, stream>>>(edge_src, cnt_u, GNN_E);

  // movie projection
  gnn_proj_relu_kernel<<<(mt + 3) / 4, 128, 0, stream>>>(movie_x, proj_w, proj_b,
                                                         movie0, mt);

  // ---- layer 1, user->movie ----
  gnn_zero_kernel<<<dmg, B, 0, stream>>>(agg_m, mN);
  gnn_scatter_add_kernel<<<sgrid, B, 0, stream>>>(user_emb, edge_src, edge_dst, agg_m, GNN_E);
  gnn_sage_linear_kernel<<<(mt + 3) / 4, 128, 0, stream>>>(
      agg_m, cnt_m, movie0, c1_um_wl, c1_um_bl, c1_um_wr, m_res, mt, 1);

  // ---- layer 1, movie->user ----
  gnn_zero_kernel<<<dug, B, 0, stream>>>(agg_u, uN);
  gnn_scatter_add_kernel<<<sgrid, B, 0, stream>>>(movie0, edge_dst, edge_src, agg_u, GNN_E);
  gnn_sage_linear_kernel<<<(ut + 3) / 4, 128, 0, stream>>>(
      agg_u, cnt_u, user_emb, c1_mu_wl, c1_mu_bl, c1_mu_wr, u_res, ut, 1);

  // ---- layer 2, user->movie ----
  gnn_zero_kernel<<<dmg, B, 0, stream>>>(agg_m, mN);
  gnn_scatter_add_kernel<<<sgrid, B, 0, stream>>>(u_res, edge_src, edge_dst, agg_m, GNN_E);
  gnn_sage_linear_kernel<<<(mt + 3) / 4, 128, 0, stream>>>(
      agg_m, cnt_m, m_res, c2_um_wl, c2_um_bl, c2_um_wr, m2, mt, 0);

  // ---- layer 2, movie->user ----
  gnn_zero_kernel<<<dug, B, 0, stream>>>(agg_u, uN);
  gnn_scatter_add_kernel<<<sgrid, B, 0, stream>>>(m_res, edge_dst, edge_src, agg_u, GNN_E);
  gnn_sage_linear_kernel<<<(ut + 3) / 4, 128, 0, stream>>>(
      agg_u, cnt_u, u_res, c2_mu_wl, c2_mu_bl, c2_mu_wr, u2, ut, 0);

  // L2 normalize
  gnn_l2norm_kernel<<<(GNN_USERS * 32 + B - 1) / B, B, 0, stream>>>(u2, GNN_USERS);
  gnn_l2norm_kernel<<<(GNN_MOVIES * 32 + B - 1) / B, B, 0, stream>>>(m2, GNN_MOVIES);

  // decoder MLP over label edges
  gnn_decoder_kernel<<<(lt + 3) / 4, 128, 0, stream>>>(
      u2, m2, lbl_src, lbl_dst, dec_w1, dec_b1, dec_w2, dec_b2, out, lt);
}
